// FDRN_39934605918325
// MI455X (gfx1250) — matrix operand order repaired and checked by reading, not compiled, not run on hardware
//
#include <hip/hip_runtime.h>
#include <hip/hip_bf16.h>

typedef _Float16 half_t;
typedef __attribute__((ext_vector_type(16))) _Float16 v16h;
typedef __attribute__((ext_vector_type(8)))  _Float16 v8h;
typedef __attribute__((ext_vector_type(8)))  float    v8f;

#define BB 16
#define LL 8192
#define LP 8194          // L + 2 halo rows (zeroed) -> branch-free conv loads
#define CC 256

// ---------------------------------------------------------------------------
// Zero the halo rows (p = -1 and p = L) of both activation buffers.
// Conv kernels only ever store interior rows, so pads stay zero all 66 passes.
// ---------------------------------------------------------------------------
__global__ __launch_bounds__(256) void zero_pads(
    half_t* __restrict__ bufH, half_t* __restrict__ bufR)
{
    int tid = blockIdx.x * 256 + threadIdx.x;   // 16 batches * 2 rows * 256 ch
    int b = tid >> 9;
    int e = (tid >> 8) & 1;                     // 0: front pad, 1: back pad
    int c = tid & 255;
    size_t row = (size_t)b * LP + (e ? (LP - 1) : 0);
    bufH[row * CC + c] = (half_t)0.f;
    bufR[row * CC + c] = (half_t)0.f;
}

// ---------------------------------------------------------------------------
// conv_in: Conv1d(1,256,k=3,'same') + ReLU -> f16 [B][LP][C] interior rows
// ---------------------------------------------------------------------------
__global__ __launch_bounds__(256) void conv_in_kernel(
    const float* __restrict__ x, const float* __restrict__ w,
    const float* __restrict__ bsc, half_t* __restrict__ out)
{
    int c = threadIdx.x;            // 256 channels
    int l = blockIdx.x;             // L
    int b = blockIdx.y;             // B
    const float* xb = x + (size_t)b * LL;
    float xm = (l > 0)      ? xb[l - 1] : 0.f;
    float xc = xb[l];
    float xp = (l < LL - 1) ? xb[l + 1] : 0.f;
    float y = w[c * 3 + 0] * xm + w[c * 3 + 1] * xc + w[c * 3 + 2] * xp + bsc[c];
    y = fmaxf(y, 0.f);
    out[((size_t)b * LP + l + 1) * CC + c] = (half_t)y;
}

// ---------------------------------------------------------------------------
// Pack W[co][ci][k] (f32) into per-lane WMMA A fragments (f16):
// Apack[cot][k][ch][lane][e], co = cot*16 + lane%16,
// ci = ch*32 + e + 16*(lane>=16)   (contiguous k order; B loads use the same)
// ---------------------------------------------------------------------------
__global__ __launch_bounds__(256) void pack_weights(
    const float* __restrict__ W, half_t* __restrict__ Apack)
{
    int idx = blockIdx.x * 256 + threadIdx.x;   // 16*3*8*32*16 = 196608
    if (idx >= 196608) return;
    int e    = idx & 15;
    int lane = (idx >> 4) & 31;
    int ch   = (idx >> 9) & 7;
    int kk   = (idx >> 12) % 3;
    int cot  = idx / 12288;
    int co = cot * 16 + (lane & 15);
    int ci = ch * 32 + e + ((lane >> 4) << 4);
    Apack[idx] = (half_t)W[((size_t)co * CC + ci) * 3 + kk];
}

// ---------------------------------------------------------------------------
// Fold BN into per-channel scale/bias:  y = conv*scale + bias
// ---------------------------------------------------------------------------
__global__ __launch_bounds__(256) void prep_sb(
    const float* b1, const float* g1, const float* be1, const float* m1, const float* v1,
    const float* b2, const float* g2, const float* be2, const float* m2, const float* v2,
    const float* bu1, float* __restrict__ sb)
{
    int c = threadIdx.x;
    float s1 = g1[c] * rsqrtf(v1[c] + 1e-5f);
    sb[c]        = s1;
    sb[256 + c]  = (b1[c] - m1[c]) * s1 + be1[c];
    float s2 = g2[c] * rsqrtf(v2[c] + 1e-5f);
    sb[512 + c]  = s2;
    sb[768 + c]  = (b2[c] - m2[c]) * s2 + be2[c];
    sb[1024 + c] = 1.0f;
    sb[1280 + c] = bu1[c];
}

// ---------------------------------------------------------------------------
// 256->256 k=3 conv as WMMA GEMM on padded f16 [B][LP][256].
// Each wave: 16(co) x 32(pos) = two D tiles sharing every A fragment.
// Inner loop is branch-free (halo rows are zero).
// ---------------------------------------------------------------------------
__global__ __launch_bounds__(256) void wmma_conv256(
    const half_t* __restrict__ in, half_t* __restrict__ out,
    const half_t* __restrict__ Apack,
    const float* __restrict__ scale, const float* __restrict__ bias,
    const half_t* __restrict__ resid, float rscale, int relu_flag)
{
    __shared__ uint4 sAq[1536];                 // 24 KB: 3 taps x 8 chunks x 512 halfs
    _Float16* sA = (_Float16*)sAq;

    const int b    = blockIdx.z;
    const int cot  = blockIdx.y;
    const int slab = blockIdx.x * 256;          // 8 waves x 32 positions
    const int tid  = threadIdx.x;

    // stage this co-tile's packed A slab into LDS (shared by all 8 waves)
    const uint4* gsrc = (const uint4*)(Apack + (size_t)cot * 12288);
    #pragma unroll
    for (int i = 0; i < 6; ++i) sAq[tid + 256 * i] = gsrc[tid + 256 * i];
    __syncthreads();

    const int wave = tid >> 5;
    const int lane = tid & 31;
    const int hi   = lane >> 4;      // K-half select (0/1)
    const int lm   = lane & 15;      // N (position) / M (row) index
    const int prow0 = slab + wave * 32 + lm;    // tile 0 position (unpadded idx)
    const int prow1 = prow0 + 16;               // tile 1

    const half_t* inb = in + (size_t)b * LP * CC;
    v8f acc0 = {}, acc1 = {};

    for (int ch = 0; ch < 8; ++ch) {
        const int ci0 = ch * 32 + hi * 16;
        if (ch < 7)  // pull next chunk's activations toward the caches
            __builtin_prefetch(inb + (size_t)(prow0 + 1) * CC + ci0 + 32, 0, 1);
        #pragma unroll
        for (int k = 0; k < 3; ++k) {
            v16h a = *(const v16h*)(sA + (((k * 8 + ch) * 32 + lane) << 4));
            // padded index: actual pos p-1+k -> row (prow + k)
            v16h bf0 = *(const v16h*)(inb + (size_t)(prow0 + k) * CC + ci0);
            v16h bf1 = *(const v16h*)(inb + (size_t)(prow1 + k) * CC + ci0);
            acc0 = __builtin_amdgcn_wmma_f32_16x16x32_f16(
                false, a, false, bf0, (short)0, acc0, false, false);
            // identical instruction, same A operand -> reuse_a hint (ISA 7.12)
            acc1 = __builtin_amdgcn_wmma_f32_16x16x32_f16(
                false, a, false, bf1, (short)0, acc1, true, false);
        }
    }

    // epilogue: lane's VGPR r holds (co = cot*16 + 8*hi + r, pos = prow)
    const int co_base = cot * 16 + hi * 8;
    float sc[8], bi[8];
    #pragma unroll
    for (int r = 0; r < 8; ++r) {
        sc[r] = scale[co_base + r];
        bi[r] = bias[co_base + r];
    }
    const size_t ob0 = ((size_t)b * LP + prow0 + 1) * CC + co_base;
    const size_t ob1 = ob0 + (size_t)16 * CC;

    float y0[8], y1[8];
    #pragma unroll
    for (int r = 0; r < 8; ++r) {
        float u0 = acc0[r] * sc[r] + bi[r];
        float u1 = acc1[r] * sc[r] + bi[r];
        if (relu_flag) { u0 = fmaxf(u0, 0.f); u1 = fmaxf(u1, 0.f); }
        y0[r] = u0; y1[r] = u1;
    }
    if (resid) {
        v8h rv0 = *(const v8h*)(resid + ob0);
        v8h rv1 = *(const v8h*)(resid + ob1);
        #pragma unroll
        for (int r = 0; r < 8; ++r) {
            y0[r] = (float)rv0[r] + rscale * y0[r];
            y1[r] = (float)rv1[r] + rscale * y1[r];
        }
    }
    v8h o0, o1;
    #pragma unroll
    for (int r = 0; r < 8; ++r) { o0[r] = (half_t)y0[r]; o1[r] = (half_t)y1[r]; }
    *(v8h*)(out + ob0) = o0;
    *(v8h*)(out + ob1) = o1;
}

// ---------------------------------------------------------------------------
// up2: Conv1d(256,4,k=3) + ReLU, pixel-shuffle [B,4,L]->[B,1,4L] + ReLU.
// Branch-free thanks to padded input.
// ---------------------------------------------------------------------------
__global__ __launch_bounds__(256) void up2_shuffle(
    const half_t* __restrict__ r, const float* __restrict__ w,
    const float* __restrict__ bsc, float* __restrict__ out)
{
    __shared__ float sw[3072];                    // w_u2 [4][256][3]
    int tid = threadIdx.x;
    #pragma unroll
    for (int i = 0; i < 12; ++i) sw[tid + 256 * i] = w[tid + 256 * i];
    __syncthreads();

    int l = blockIdx.x * 256 + tid;
    int b = blockIdx.y;
    float a0 = bsc[0], a1 = bsc[1], a2 = bsc[2], a3 = bsc[3];
    const half_t* rb = r + (size_t)b * LP * CC;
    #pragma unroll
    for (int k = 0; k < 3; ++k) {
        const half_t* rp = rb + (size_t)(l + k) * CC;   // padded row, halo = 0
        for (int ci = 0; ci < 256; ci += 4) {
            float x0 = (float)rp[ci],     x1 = (float)rp[ci + 1];
            float x2 = (float)rp[ci + 2], x3 = (float)rp[ci + 3];
            #pragma unroll
            for (int j = 0; j < 4; ++j) {
                const float* wc = sw + (j * 256 + ci) * 3 + k;
                float* ac = (j == 0) ? &a0 : (j == 1) ? &a1 : (j == 2) ? &a2 : &a3;
                *ac += wc[0] * x0 + wc[3] * x1 + wc[6] * x2 + wc[9] * x3;
            }
        }
    }
    float4 o = { fmaxf(a0, 0.f), fmaxf(a1, 0.f), fmaxf(a2, 0.f), fmaxf(a3, 0.f) };
    *(float4*)(out + ((size_t)b * LL + l) * 4) = o;
}

// ---------------------------------------------------------------------------
extern "C" void kernel_launch(void* const* d_in, const int* in_sizes, int n_in,
                              void* d_out, int out_size, void* d_ws, size_t ws_size,
                              hipStream_t stream)
{
    const float* x    = (const float*)d_in[0];
    const float* w_in = (const float*)d_in[3];
    const float* b_in = (const float*)d_in[4];
    const float* w_r1 = (const float*)d_in[5];
    const float* b_r1 = (const float*)d_in[6];
    const float* g1   = (const float*)d_in[7];
    const float* be1  = (const float*)d_in[8];
    const float* m1   = (const float*)d_in[9];
    const float* v1   = (const float*)d_in[10];
    const float* w_r2 = (const float*)d_in[11];
    const float* b_r2 = (const float*)d_in[12];
    const float* g2   = (const float*)d_in[13];
    const float* be2  = (const float*)d_in[14];
    const float* m2   = (const float*)d_in[15];
    const float* v2   = (const float*)d_in[16];
    const float* w_u1 = (const float*)d_in[17];
    const float* b_u1 = (const float*)d_in[18];
    const float* w_u2 = (const float*)d_in[19];
    const float* b_u2 = (const float*)d_in[20];

    // workspace layout (f16 activations, both buffers L2-resident)
    const size_t BUF = (size_t)BB * LP * CC * sizeof(half_t);   // ~64 MiB each
    const size_t APK = 196608 * sizeof(half_t);                 // 384 KiB each
    const size_t NEED = 2 * BUF + 3 * APK + 1536 * sizeof(float);
    if (ws_size < NEED) return;

    char* ws = (char*)d_ws;
    half_t* bufH = (half_t*)ws;
    half_t* bufR = (half_t*)(ws + BUF);
    half_t* A1   = (half_t*)(ws + 2 * BUF);
    half_t* A2   = A1 + 196608;
    half_t* AU   = A2 + 196608;
    float*  sb   = (float*)(AU + 196608);

    zero_pads<<<32, 256, 0, stream>>>(bufH, bufR);
    conv_in_kernel<<<dim3(LL, BB), 256, 0, stream>>>(x, w_in, b_in, bufH);
    pack_weights<<<768, 256, 0, stream>>>(w_r1, A1);
    pack_weights<<<768, 256, 0, stream>>>(w_r2, A2);
    pack_weights<<<768, 256, 0, stream>>>(w_u1, AU);
    prep_sb<<<1, 256, 0, stream>>>(b_r1, g1, be1, m1, v1,
                                   b_r2, g2, be2, m2, v2, b_u1, sb);

    dim3 cg(LL / 256, 16, BB);   // 32 pos-slabs x 16 co-tiles x 16 batches
    for (int i = 0; i < 32; ++i) {
        // r = relu(bn1(conv1(h)))
        wmma_conv256<<<cg, 256, 0, stream>>>(bufH, bufR, A1, sb, sb + 256,
                                             nullptr, 0.f, 1);
        // h = h + 0.15 * bn2(conv2(r))   (in-place on h: own-element RMW only)
        wmma_conv256<<<cg, 256, 0, stream>>>(bufR, bufH, A2, sb + 512, sb + 768,
                                             bufH, 0.15f, 0);
    }
    // up1: relu(conv(h))
    wmma_conv256<<<cg, 256, 0, stream>>>(bufH, bufR, AU, sb + 1024, sb + 1280,
                                         nullptr, 0.f, 1);
    // up2 + pixel shuffle + relu
    up2_shuffle<<<dim3(LL / 256, BB), 256, 0, stream>>>(bufR, w_u2, b_u2,
                                                        (float*)d_out);
}
